// HPUMLAImpl_64561948394281
// MI455X (gfx1250) — matrix operand order repaired and checked by reading, not compiled, not run on hardware
//
#include <hip/hip_runtime.h>
#include <hip/hip_bf16.h>
#include <math.h>

typedef __attribute__((ext_vector_type(16))) __bf16 bf16x16;
typedef __attribute__((ext_vector_type(8)))  float  f32x8;

// ---------------- WMMA fragment helpers (CDNA5 16x16x32 bf16, wave32) -------

__device__ __forceinline__ f32x8 wmma_bf16(bf16x16 a, bf16x16 b, f32x8 c) {
    return __builtin_amdgcn_wmma_f32_16x16x32_bf16(false, a, false, b, (short)0, c,
                                                   false, false);
}

// A tile 16x32 (MxK), row-major source, base = &A[row0*ld + k0], ld mult of 8.
// Lane l (l<16): M=l, K in {kh..kh+7, 16+kh..16+kh+7} with kh = (l>>4)*8.
__device__ __forceinline__ bf16x16 load_a16x32(const __bf16* base, int ld, int lane) {
    int m  = lane & 15;
    int kh = (lane >> 4) << 3;                  // 0 or 8
    const __bf16* p = base + m * ld + kh;
    union { bf16x16 v; uint4 u[2]; } r;
    r.u[0] = *reinterpret_cast<const uint4*>(p);        // K kh..kh+7
    r.u[1] = *reinterpret_cast<const uint4*>(p + 16);   // K 16+kh..16+kh+7
    return r.v;
}

// B tile 32x16 (KxN), column-major source (column n contiguous in K):
// base = &Bt[n0*ld + k0]; lane: N=lane&15, K in {kh..kh+7, 16+kh..16+kh+7} with kh = (lane>>4)*8
__device__ __forceinline__ bf16x16 load_b32x16(const __bf16* base, int ld, int lane) {
    int n  = lane & 15;
    int kh = (lane >> 4) << 3;                  // 0 or 8
    const __bf16* p = base + n * ld + kh;
    union { bf16x16 v; uint4 u[2]; } r;
    r.u[0] = *reinterpret_cast<const uint4*>(p);
    r.u[1] = *reinterpret_cast<const uint4*>(p + 16);
    return r.v;
}

// ---------------- conversion kernels ----------------------------------------

__global__ void convert_bf16(const float* __restrict__ in, __bf16* __restrict__ out,
                             long n) {
    long stride = (long)gridDim.x * blockDim.x;
    for (long i = blockIdx.x * (long)blockDim.x + threadIdx.x; i < n; i += stride)
        out[i] = (__bf16)in[i];
}

// in: [batch][R][C] f32 row-major -> out: [batch][C][R] bf16 (col-major of in)
__global__ void transpose_bf16(const float* __restrict__ in, __bf16* __restrict__ out,
                               int R, int C, int nBatch) {
    long total = (long)nBatch * R * C;
    long stride = (long)gridDim.x * blockDim.x;
    for (long id = blockIdx.x * (long)blockDim.x + threadIdx.x; id < total; id += stride) {
        long b   = id / ((long)R * C);
        long rem = id - b * (long)R * C;
        int r = (int)(rem / C);
        int c = (int)(rem % C);
        out[b * (long)R * C + (long)c * R + r] = (__bf16)in[id];
    }
}

__global__ void zero_f32(float* p, long n) {
    long stride = (long)gridDim.x * blockDim.x;
    for (long i = blockIdx.x * (long)blockDim.x + threadIdx.x; i < n; i += stride)
        p[i] = 0.0f;
}

// ---------------- generic batched WMMA GEMM ---------------------------------
// C[M,N] = A[M,K] (bf16 row-major, ld=lda) * Bt (bf16, column-major, ld=ldb)
// blockIdx.y = batch; one 16x16 tile per wave.
__global__ void wmma_gemm(const __bf16* __restrict__ A, int lda, long aStride,
                          const __bf16* __restrict__ Bt, int ldb, long bStride,
                          float* Cf, __bf16* Cb, int ldc, long cStride,
                          int M, int N, int K, float scale) {
    int lane = threadIdx.x & 31;
    int wid  = blockIdx.x * (blockDim.x >> 5) + (threadIdx.x >> 5);
    int tn = N >> 4, tm = M >> 4;
    if (wid >= tm * tn) return;
    int tmi = wid / tn, tni = wid % tn;
    long batch = blockIdx.y;
    const __bf16* Ab = A  + batch * aStride + (long)tmi * 16 * lda;
    const __bf16* Bb = Bt + batch * bStride + (long)tni * 16 * ldb;
    f32x8 acc = {};
    for (int k0 = 0; k0 < K; k0 += 32) {
        bf16x16 a = load_a16x32(Ab + k0, lda, lane);
        bf16x16 b = load_b32x16(Bb + k0, ldb, lane);
        acc = wmma_bf16(a, b, acc);
    }
    int n  = lane & 15;
    int mh = (lane >> 4) << 3;
    long row0 = (long)tmi * 16 + mh;
    long col  = (long)tni * 16 + n;
    if (Cf) {
        float* p = Cf + batch * cStride + row0 * ldc + col;
#pragma unroll
        for (int i = 0; i < 8; ++i) p[(long)i * ldc] = acc[i] * scale;
    }
    if (Cb) {
        __bf16* p = Cb + batch * cStride + row0 * ldc + col;
#pragma unroll
        for (int i = 0; i < 8; ++i) p[(long)i * ldc] = (__bf16)(acc[i] * scale);
    }
}

// ---------------- RoPE + cache-update side buffers --------------------------
// grid = B, block = 1024.  q_bf[b][h][576]: [0:512) nope (written by GEMM3),
// [512:576) rope part written here (scaled).
__global__ void rope_update(const float* __restrict__ qpe, const float* __restrict__ k_pe,
                            const float* __restrict__ k_c, const int* __restrict__ pos,
                            __bf16* __restrict__ q_bf, float* __restrict__ upd_k,
                            float* __restrict__ upd_v, float scale) {
    int b   = blockIdx.x;
    int tid = threadIdx.x;
    float posf = (float)pos[b];
    int h = tid >> 5, i = tid & 31;                 // 32 heads x 32 pairs
    float inv = powf(10000.0f, -(float)i / 32.0f);
    float ang = posf * inv;
    float s, c;
    sincosf(ang, &s, &c);
    float x1 = qpe[b * 2048 + h * 64 + i];
    float x2 = qpe[b * 2048 + h * 64 + i + 32];
    q_bf[(long)b * 18432 + h * 576 + 512 + i]      = (__bf16)((x1 * c - x2 * s) * scale);
    q_bf[(long)b * 18432 + h * 576 + 512 + 32 + i] = (__bf16)((x2 * c + x1 * s) * scale);
    if (h == 0) {
        float k1 = k_pe[b * 64 + i], k2 = k_pe[b * 64 + i + 32];
        upd_k[b * 64 + i]      = k1 * c - k2 * s;
        upd_k[b * 64 + i + 32] = k2 * c + k1 * s;
    }
    for (int l = tid; l < 512; l += blockDim.x)
        upd_v[b * 512 + l] = k_c[b * 512 + l];
}

__global__ void ov_init(int* ov, int n) {
    int i = blockIdx.x * blockDim.x + threadIdx.x;
    if (i < n) ov[i] = -1;
}
__global__ void ov_set(int* ov, const int* bidx, const int* boff, int nb) {
    int g = blockIdx.x * blockDim.x + threadIdx.x;
    if (g < nb) ov[bidx[g]] = (g << 8) | (boff[g] & 255);
}

// ---------------- attention pass A: logits ----------------------------------
#define CHUNK 192
#define KLD   200   // padded row stride (mult of 8 -> 16B aligned b128 LDS reads)
__global__ void attn_logits(const float* __restrict__ cache_k, const float* __restrict__ cache_v,
                            const __bf16* __restrict__ q_bf, const int* __restrict__ block_list,
                            const int* __restrict__ block_groups, const float* __restrict__ bias,
                            const int* __restrict__ ov, const float* __restrict__ upd_k,
                            const float* __restrict__ upd_v, float* __restrict__ logits) {
    __shared__ __align__(16) __bf16 kc[128 * KLD];
    int n   = blockIdx.x;
    int bid = block_list[n];
    int g   = block_groups[n];
    int tid = threadIdx.x, lane = tid & 31, w = tid >> 5;   // 8 waves
    int ovv    = ov[bid];
    int ov_row = (ovv < 0) ? -1 : (ovv & 255);
    int ov_g   = ovv >> 8;
    const __bf16* qg = q_bf + (long)g * 18432;
    f32x8 acc0 = {}, acc1 = {};
    for (int c0 = 0; c0 < 576; c0 += CHUNK) {
        __syncthreads();
        for (int idx = tid; idx < 128 * CHUNK; idx += blockDim.x) {
            int tok = idx / CHUNK, d = idx % CHUNK;
            int dd  = c0 + d;
            float v;
            if (tok == ov_row)
                v = (dd < 512) ? upd_v[ov_g * 512 + dd] : upd_k[ov_g * 64 + (dd - 512)];
            else
                v = (dd < 512) ? cache_v[(long)bid * 65536 + tok * 512 + dd]
                               : cache_k[(long)bid * 8192 + tok * 64 + (dd - 512)];
            kc[tok * KLD + d] = (__bf16)v;
        }
        __syncthreads();
        const __bf16* bbase = kc + (w * 16) * KLD;          // wave's 16 tokens
        for (int kk = 0; kk < CHUNK; kk += 32) {
            bf16x16 b  = load_b32x16(bbase + kk, KLD, lane);
            bf16x16 a0 = load_a16x32(qg + c0 + kk, 576, lane);           // heads 0..15
            bf16x16 a1 = load_a16x32(qg + 16 * 576 + c0 + kk, 576, lane); // heads 16..31
            acc0 = wmma_bf16(a0, b, acc0);
            acc1 = wmma_bf16(a1, b, acc1);
        }
    }
    int ncol = lane & 15, mh = (lane >> 4) << 3;
    int tok  = w * 16 + ncol;
    float bv = bias[(long)n * 128 + tok];
    float* lp = logits + (long)n * 4096 + tok;
#pragma unroll
    for (int i = 0; i < 8; ++i) {
        lp[(mh + i) * 128]      = acc0[i] + bv;
        lp[(16 + mh + i) * 128] = acc1[i] + bv;
    }
}

// ---------------- segment max over groups (general via block_groups) --------
__global__ void reduce_gmax(const float* __restrict__ logits, const int* __restrict__ bg,
                            float* __restrict__ gmax, int NB) {
    int wid  = (blockIdx.x * blockDim.x + threadIdx.x) >> 5;
    int lane = threadIdx.x & 31;
    if (wid >= 32 * 32) return;
    int b = wid >> 5, h = wid & 31;
    float m = -1e30f;
    for (int n = 0; n < NB; ++n) {
        if (bg[n] != b) continue;
        const float* base = logits + (long)n * 4096 + h * 128;
        for (int t = lane; t < 128; t += 32) m = fmaxf(m, base[t]);
    }
    for (int off = 16; off > 0; off >>= 1) m = fmaxf(m, __shfl_xor(m, off, 32));
    if (lane == 0) gmax[wid] = m;
}

// ---------------- attention pass B: p = exp(l-gmax), p @ V ------------------
#define VLD 136
__global__ void attn_av(const float* __restrict__ cache_v, const float* __restrict__ logits,
                        const float* __restrict__ gmax, const int* __restrict__ block_list,
                        const int* __restrict__ block_groups, const int* __restrict__ ov,
                        const float* __restrict__ upd_v, float* __restrict__ num,
                        float* __restrict__ den) {
    __shared__ __align__(16) __bf16 pbf[32 * 128];
    __shared__ __align__(16) __bf16 vt[128 * VLD];
    int n   = blockIdx.x;
    int bid = block_list[n];
    int g   = block_groups[n];
    int tid = threadIdx.x, lane = tid & 31, w = tid >> 5;
    int ovv    = ov[bid];
    int ov_row = (ovv < 0) ? -1 : (ovv & 255);
    int ov_g   = ovv >> 8;
    {   // stage p (bf16) + accumulate den
        int base = tid * 16;                 // 256*16 = 4096 = 32h * 128tok
        int h    = base >> 7;
        float gm = gmax[g * 32 + h];
        float s  = 0.0f;
#pragma unroll
        for (int i = 0; i < 16; ++i) {
            float p = expf(logits[(long)n * 4096 + base + i] - gm);
            s += p;
            pbf[base + i] = (__bf16)p;
        }
        atomicAdd(&den[g * 32 + h], s);
    }
    for (int l0 = 0; l0 < 512; l0 += 128) {
        __syncthreads();
        for (int idx = tid; idx < 128 * 128; idx += blockDim.x) {
            int tok = idx >> 7, l = idx & 127;
            float v = (tok == ov_row) ? upd_v[ov_g * 512 + l0 + l]
                                      : cache_v[(long)bid * 65536 + tok * 512 + l0 + l];
            vt[l * VLD + tok] = (__bf16)v;   // transposed: column l contiguous in tok
        }
        __syncthreads();
        f32x8 acc0 = {}, acc1 = {};
        const __bf16* bbase = vt + (w * 16) * VLD;
        for (int kk = 0; kk < 128; kk += 32) {
            bf16x16 b  = load_b32x16(bbase + kk, VLD, lane);
            bf16x16 a0 = load_a16x32(pbf + kk, 128, lane);
            bf16x16 a1 = load_a16x32(pbf + 16 * 128 + kk, 128, lane);
            acc0 = wmma_bf16(a0, b, acc0);
            acc1 = wmma_bf16(a1, b, acc1);
        }
        int ncol = lane & 15, mh = (lane >> 4) << 3;
        int l = l0 + w * 16 + ncol;
#pragma unroll
        for (int i = 0; i < 8; ++i) {
            atomicAdd(&num[(long)g * 16384 + (mh + i) * 512 + l], acc0[i]);
            atomicAdd(&num[(long)g * 16384 + (16 + mh + i) * 512 + l], acc1[i]);
        }
    }
}

__global__ void attn_norm(const float* __restrict__ num, const float* __restrict__ den,
                          __bf16* __restrict__ attn_bf, long n) {
    long stride = (long)gridDim.x * blockDim.x;
    for (long i = blockIdx.x * (long)blockDim.x + threadIdx.x; i < n; i += stride) {
        long bh = i >> 9;                    // num layout [b][h][512]
        attn_bf[i] = (__bf16)(num[i] / den[bh]);
    }
}

// ---------------- launch ----------------------------------------------------

extern "C" void kernel_launch(void* const* d_in, const int* in_sizes, int n_in,
                              void* d_out, int out_size, void* d_ws, size_t ws_size,
                              hipStream_t stream) {
    const float* hidden = (const float*)d_in[0];   // 32x1536
    const float* k_c    = (const float*)d_in[1];   // 32x1x512
    const float* k_pe   = (const float*)d_in[2];   // 32x1x64
    const float* ck     = (const float*)d_in[3];   // cache_k [CB][128][64]
    const float* cv     = (const float*)d_in[4];   // cache_v [CB][128][512]
    const float* W_Q    = (const float*)d_in[5];   // 1536x4096
    const float* W_UK   = (const float*)d_in[6];   // 32x128x512
    const float* W_QR   = (const float*)d_in[7];   // 1536x2048
    const float* W_UV   = (const float*)d_in[8];   // 32x512x128
    const float* W_O    = (const float*)d_in[9];   // 4096x5120
    const int*   pos    = (const int*)d_in[10];
    const int*   blist  = (const int*)d_in[11];
    const int*   bgrp   = (const int*)d_in[12];
    const float* bbias  = (const float*)d_in[13];  // [NB][128]
    const int*   bidx   = (const int*)d_in[14];
    const int*   boff   = (const int*)d_in[15];
    // d_in[16] = block_mapping (one-hot of block_groups; unused)
    float* out = (float*)d_out;

    const int NB     = in_sizes[11];
    const int nCache = in_sizes[3] / (128 * 64);
    const float scale = 0.07216878364870322f;      // 1/sqrt(128+64)

    char* ws = (char*)d_ws;
    size_t o = 0;
    auto alloc = [&](size_t bytes) -> char* {
        char* p = ws + o;
        o += (bytes + 255) & ~(size_t)255;
        return p;
    };
    __bf16* WQt   = (__bf16*)alloc((size_t)4096 * 1536 * 2);
    __bf16* WQRt  = (__bf16*)alloc((size_t)2048 * 1536 * 2);
    __bf16* WUKt  = (__bf16*)alloc((size_t)32 * 512 * 128 * 2);
    __bf16* WUVt  = (__bf16*)alloc((size_t)32 * 128 * 512 * 2);
    __bf16* WOt   = (__bf16*)alloc((size_t)5120 * 4096 * 2);
    __bf16* qc_bf = (__bf16*)alloc((size_t)32 * 1536 * 2);
    __bf16* q1_bf = (__bf16*)alloc((size_t)32 * 4096 * 2);
    float*  qpe   = (float*) alloc((size_t)32 * 2048 * 4);
    __bf16* q_bf  = (__bf16*)alloc((size_t)32 * 18432 * 2);
    float*  upd_k = (float*) alloc((size_t)32 * 64 * 4);
    float*  upd_v = (float*) alloc((size_t)32 * 512 * 4);
    int*    ovmap = (int*)   alloc((size_t)nCache * 4);
    float*  logit = (float*) alloc((size_t)NB * 4096 * 4);
    float*  gmx   = (float*) alloc((size_t)1024 * 4);
    float*  num   = (float*) alloc((size_t)32 * 16384 * 4);
    float*  den   = (float*) alloc((size_t)1024 * 4);
    __bf16* attnb = (__bf16*)alloc((size_t)32 * 16384 * 2);
    __bf16* omid  = (__bf16*)alloc((size_t)32 * 4096 * 2);

    // --- weight / activation conversion (bf16, B-operands transposed) ---
    convert_bf16<<<192, 256, 0, stream>>>(hidden, qc_bf, 32L * 1536);
    transpose_bf16<<<8192, 256, 0, stream>>>(W_Q,  WQt,  1536, 4096, 1);
    transpose_bf16<<<4096, 256, 0, stream>>>(W_QR, WQRt, 1536, 2048, 1);
    transpose_bf16<<<4096, 256, 0, stream>>>(W_UK, WUKt, 128, 512, 32);
    transpose_bf16<<<4096, 256, 0, stream>>>(W_UV, WUVt, 512, 128, 32);
    transpose_bf16<<<32768, 256, 0, stream>>>(W_O, WOt, 4096, 5120, 1);

    // --- q projections ---
    // q1 = hidden @ W_Q  (32x4096)
    wmma_gemm<<<dim3(64, 1), 256, 0, stream>>>(qc_bf, 1536, 0, WQt, 1536, 0,
                                               nullptr, q1_bf, 4096, 0,
                                               32, 4096, 1536, 1.0f);
    // qpe = hidden @ W_QR  (32x2048, f32 for rope)
    wmma_gemm<<<dim3(32, 1), 256, 0, stream>>>(qc_bf, 1536, 0, WQRt, 1536, 0,
                                               qpe, nullptr, 2048, 0,
                                               32, 2048, 1536, 1.0f);
    // q_nope[b,h,:] = q1[b,h,:] @ W_UK[h]  -> q_bf[b][h][0:512], scaled
    wmma_gemm<<<dim3(8, 32), 256, 0, stream>>>(q1_bf, 4096, 128, WUKt, 128, 65536,
                                               nullptr, q_bf, 18432, 576,
                                               32, 512, 128, scale);
    // rope(q_pe) -> q_bf[b][h][512:576]; rope(k_pe)->upd_k; k_c->upd_v
    rope_update<<<32, 1024, 0, stream>>>(qpe, k_pe, k_c, pos, q_bf, upd_k, upd_v, scale);

    // --- virtual cache update map ---
    ov_init<<<(nCache + 255) / 256, 256, 0, stream>>>(ovmap, nCache);
    ov_set<<<1, 256, 0, stream>>>(ovmap, bidx, boff, 32);

    // --- attention ---
    zero_f32<<<2048, 256, 0, stream>>>(num, 32L * 16384);
    zero_f32<<<4, 256, 0, stream>>>(den, 1024);
    attn_logits<<<NB, 256, 0, stream>>>(ck, cv, q_bf, blist, bgrp, bbias,
                                        ovmap, upd_k, upd_v, logit);
    reduce_gmax<<<128, 256, 0, stream>>>(logit, bgrp, gmx, NB);
    attn_av<<<NB, 256, 0, stream>>>(cv, logit, gmx, blist, bgrp, ovmap, upd_v, num, den);
    attn_norm<<<2048, 256, 0, stream>>>(num, den, attnb, 32L * 16384);

    // --- output projections ---
    // omid[b][h*128+v] = attn[b,h,:] @ W_UV[h]
    wmma_gemm<<<dim3(2, 32), 256, 0, stream>>>(attnb, 16384, 512, WUVt, 512, 65536,
                                               nullptr, omid, 4096, 128,
                                               32, 128, 512, 1.0f);
    // out = omid @ W_O   (32x5120, f32)
    wmma_gemm<<<dim3(80, 1), 256, 0, stream>>>(omid, 4096, 0, WOt, 4096, 0,
                                               out, nullptr, 5120, 0,
                                               32, 5120, 4096, 1.0f);
    (void)n_in; (void)out_size; (void)ws_size;
}
